// Hi_Patch_7593502179555
// MI455X (gfx1250) — matrix operand order repaired and checked by reading, not compiled, not run on hardware
//
#include <hip/hip_runtime.h>
#include <hip/hip_bf16.h>
#include <float.h>

typedef __attribute__((ext_vector_type(16))) _Float16 v16h;
typedef __attribute__((ext_vector_type(8)))  _Float16 v8h;
typedef __attribute__((ext_vector_type(4)))  _Float16 v4h;
typedef __attribute__((ext_vector_type(8)))  float    v8f;

#define N_NODES 40000
#define E_EDGES 640000
#define D_DIM   128
#define H_HEADS 4
#define DKK     32
#define W_HALVES (H_HEADS*3*DKK*D_DIM)   // 49152 halves per weight tensor
#define SLICE_H  (3*DKK*D_DIM)           // 12288 halves per head slice
#define LS      136                      // padded LDS row stride (halves): 272B -> 4-bank skew
#define INV_SQRT_DK 0.17677669529663687f // 1/sqrt(32)
#define LOG_ALPHA  (-0.10536051565782628f) // ln(0.9)

// ---- sortable-uint float keys for deterministic atomic max ----
__device__ __forceinline__ unsigned fkey(float f) {
  unsigned u = __float_as_uint(f);
  return (u & 0x80000000u) ? ~u : (u | 0x80000000u);
}
__device__ __forceinline__ float funkey(unsigned k) {
  unsigned u = (k & 0x80000000u) ? (k & 0x7FFFFFFFu) : ~k;
  return __uint_as_float(u);
}

// ---- A fragment (16-bit A 16x32 layout): lane<16 holds K {0..7,16..23},
//      lane>=16 holds K {8..15,24..31} of the 32-wide chunk ----
__device__ __forceinline__ v16h load_a(const _Float16* __restrict__ row, int kc, int hi8) {
  const v8h lo = *reinterpret_cast<const v8h*>(row + kc + hi8);
  const v8h hi = *reinterpret_cast<const v8h*>(row + kc + 16 + hi8);
  v16h a;
#pragma unroll
  for (int i = 0; i < 8; ++i) { a[i] = lo[i]; a[i + 8] = hi[i]; }
  return a;
}

// ---- B fragment from (16B-aligned) padded LDS: halves p[0..7] and p[16..23] ----
__device__ __forceinline__ v16h load_b(const _Float16* p) {
  const v8h lo = *reinterpret_cast<const v8h*>(p);
  const v8h hi = *reinterpret_cast<const v8h*>(p + 16);
  v16h b;
#pragma unroll
  for (int i = 0; i < 8; ++i) { b[i] = lo[i]; b[i + 8] = hi[i]; }
  return b;
}

// ===================== K1: LayerNorm -> f16 =====================
__global__ __launch_bounds__(256)
void k_layernorm(const float* __restrict__ x, const float* __restrict__ gamma,
                 const float* __restrict__ beta, _Float16* __restrict__ xn16) {
  const int lane = threadIdx.x & 31;
  const int row  = blockIdx.x * 8 + (threadIdx.x >> 5);
  const float4 v = *reinterpret_cast<const float4*>(x + (size_t)row * D_DIM + lane * 4);
  float s = v.x + v.y + v.z + v.w;
#pragma unroll
  for (int m = 16; m >= 1; m >>= 1) s += __shfl_xor(s, m, 32);
  const float mu = s * (1.0f / 128.0f);
  const float d0 = v.x - mu, d1 = v.y - mu, d2 = v.z - mu, d3 = v.w - mu;
  float q = d0 * d0 + d1 * d1 + d2 * d2 + d3 * d3;
#pragma unroll
  for (int m = 16; m >= 1; m >>= 1) q += __shfl_xor(q, m, 32);
  const float rstd = rsqrtf(q * (1.0f / 128.0f) + 1e-5f);
  const float4 g = *reinterpret_cast<const float4*>(gamma + lane * 4);
  const float4 b = *reinterpret_cast<const float4*>(beta + lane * 4);
  v4h o;
  o[0] = (_Float16)(d0 * rstd * g.x + b.x);
  o[1] = (_Float16)(d1 * rstd * g.y + b.y);
  o[2] = (_Float16)(d2 * rstd * g.z + b.z);
  o[3] = (_Float16)(d3 * rstd * g.w + b.w);
  *reinterpret_cast<v4h*>(xn16 + (size_t)row * D_DIM + lane * 4) = o;
}

// ===================== K2: weights f32 (D x DK) -> f16 col-major (DK x D) ==
__global__ void k_prep_w(const float* __restrict__ Wq, const float* __restrict__ Wk,
                         const float* __restrict__ Wv, _Float16* __restrict__ q16,
                         _Float16* __restrict__ k16, _Float16* __restrict__ v16) {
  const int i = blockIdx.x * blockDim.x + threadIdx.x;
  if (i >= 3 * W_HALVES) return;
  const int which = i / W_HALVES;
  const int o = i % W_HALVES;
  const int k  = o & 127;          // K (input dim)
  const int n  = (o >> 7) & 31;    // output column
  const int ht = o >> 12;          // h*3+t
  const int in = (ht * 128 + k) * 32 + n;
  const float* W = (which == 0) ? Wq : (which == 1) ? Wk : Wv;
  _Float16*   O  = (which == 0) ? q16 : (which == 1) ? k16 : v16;
  O[o] = (_Float16)W[in];
}

// ===================== K3: per-edge Q,K via WMMA; att + per-head max =======
__global__ __launch_bounds__(256)
void k_edge_qk(const _Float16* __restrict__ xn16, const _Float16* __restrict__ Wq16,
               const _Float16* __restrict__ Wk16, const float* __restrict__ bq,
               const float* __restrict__ bk, const int* __restrict__ ei,
               const float* __restrict__ ev, const float* __restrict__ m1,
               const float* __restrict__ m2, float* __restrict__ att,
               unsigned int* __restrict__ maxkeys) {
  __shared__ __align__(16) _Float16 sWq[96 * LS];
  __shared__ __align__(16) _Float16 sWk[96 * LS];
  const int lane = threadIdx.x & 31;
  const int tile = blockIdx.x * 8 + (threadIdx.x >> 5);
  const int col  = lane & 15;
  const int half = lane >> 4;
  const int hi8  = half * 8;
  const int e    = tile * 16 + col;

  const int srcn = ei[e];
  const int dstn = ei[E_EDGES + e];
  int t = 0;
  if (m1[e] > 0.5f) t = 1;
  if (m2[e] > 0.5f) t = 2;
  const float dec = expf(LOG_ALPHA * fabsf(ev[e]));

  const _Float16* qrow = xn16 + (size_t)dstn * D_DIM; // q uses xi = xn[dst]
  const _Float16* krow = xn16 + (size_t)srcn * D_DIM; // k uses xj = xn[src]

  // head-invariant A fragments: each lane's full row as 4 K-chunks
  v16h aq[4], ak[4];
#pragma unroll
  for (int j = 0; j < 4; ++j) {
    aq[j] = load_a(qrow, j * 32, hi8);
    ak[j] = load_a(krow, j * 32, hi8);
  }

  for (int h = 0; h < H_HEADS; ++h) {
    // stage this head's Q/K weight slices (3 types x 32 cols x 128 K) in LDS
    __syncthreads();
    for (int i = threadIdx.x; i < 1536; i += 256) {
      const int row = i >> 4, piece = (i & 15) * 8;
      *reinterpret_cast<float4*>(sWq + row * LS + piece) =
          *reinterpret_cast<const float4*>(Wq16 + (size_t)h * SLICE_H + row * 128 + piece);
      *reinterpret_cast<float4*>(sWk + row * LS + piece) =
          *reinterpret_cast<const float4*>(Wk16 + (size_t)h * SLICE_H + row * 128 + piece);
    }
    __syncthreads();

    v8f cq0 = {}, cq1 = {}, ck0 = {}, ck1 = {};
#pragma unroll
    for (int c = 0; c < 12; ++c) {
      const int tc = c >> 2, kc = (c & 3) * 32;
      if (__ballot(t == tc)) {
        v16h a_q = {}, a_k = {};
        if (t == tc) { a_q = aq[c & 3]; a_k = ak[c & 3]; }
        const int wb0 = (tc * 32 + col) * LS + kc + half * 8;
        const int wb1 = (tc * 32 + 16 + col) * LS + kc + half * 8;
        cq0 = __builtin_amdgcn_wmma_f32_16x16x32_f16(false, a_q, false, load_b(sWq + wb0),
                                                     (short)0, cq0, false, false);
        cq1 = __builtin_amdgcn_wmma_f32_16x16x32_f16(false, a_q, false, load_b(sWq + wb1),
                                                     (short)0, cq1, false, false);
        ck0 = __builtin_amdgcn_wmma_f32_16x16x32_f16(false, a_k, false, load_b(sWk + wb0),
                                                     (short)0, ck0, false, false);
        ck1 = __builtin_amdgcn_wmma_f32_16x16x32_f16(false, a_k, false, load_b(sWk + wb1),
                                                     (short)0, ck1, false, false);
      }
    }

    float tmax = -FLT_MAX;
#pragma unroll
    for (int r = 0; r < 8; ++r) {
      const int srow = r + hi8;                 // edge row this half's vgpr r holds
      const int srcl = (half << 4) + srow;      // lane owning that edge's metadata
      const int   tr = __shfl(t, srcl, 32);
      const float dr = __shfl(dec, srcl, 32);
      const float qv0 = cq0[r] + bq[(h * 3 + tr) * 32 + col];
      const float qv1 = cq1[r] + bq[(h * 3 + tr) * 32 + 16 + col];
      const float kv0 = ck0[r] + bk[(h * 3 + tr) * 32 + col];
      const float kv1 = ck1[r] + bk[(h * 3 + tr) * 32 + 16 + col];
      float p = qv0 * kv0 + qv1 * kv1;
      p += __shfl_xor(p, 1, 32); p += __shfl_xor(p, 2, 32);
      p += __shfl_xor(p, 4, 32); p += __shfl_xor(p, 8, 32);
      const float a = p * INV_SQRT_DK * dr;
      if (col == 0) att[(size_t)(tile * 16 + srow) * H_HEADS + h] = a;
      tmax = fmaxf(tmax, a);
    }
    tmax = fmaxf(tmax, __shfl_xor(tmax, 16, 32));
    if (lane == 0) atomicMax(maxkeys + h, fkey(tmax));
  }
}

// ===================== K4: exp(att - max), per-dst denominator =============
__global__ void k_softmax_denom(float* __restrict__ att, const int* __restrict__ ei,
                                const unsigned int* __restrict__ maxkeys,
                                float* __restrict__ denom) {
  const int i = blockIdx.x * blockDim.x + threadIdx.x;
  if (i >= E_EDGES * H_HEADS) return;
  const int e = i >> 2, h = i & 3;
  const float a = expf(att[i] - funkey(maxkeys[h]));
  att[i] = a;
  atomicAdd(denom + (size_t)ei[E_EDGES + e] * H_HEADS + h, a);
}

// ===================== K5: per-edge V via WMMA; scatter messages ===========
__global__ __launch_bounds__(256)
void k_edge_v(const _Float16* __restrict__ xn16, const _Float16* __restrict__ Wv16,
              const float* __restrict__ bv, const int* __restrict__ ei,
              const float* __restrict__ m1, const float* __restrict__ m2,
              const float* __restrict__ att, const float* __restrict__ denom,
              float* __restrict__ aggr) {
  __shared__ __align__(16) _Float16 sWv[96 * LS];
  const int lane = threadIdx.x & 31;
  const int tile = blockIdx.x * 8 + (threadIdx.x >> 5);
  const int col  = lane & 15;
  const int half = lane >> 4;
  const int hi8  = half * 8;
  const int e    = tile * 16 + col;

  const int srcn = ei[e];
  const int dstn = ei[E_EDGES + e];
  int t = 0;
  if (m1[e] > 0.5f) t = 1;
  if (m2[e] > 0.5f) t = 2;

  const _Float16* vrow = xn16 + (size_t)srcn * D_DIM; // v uses xj = xn[src]
  v16h av[4];
#pragma unroll
  for (int j = 0; j < 4; ++j) av[j] = load_a(vrow, j * 32, hi8);

  for (int h = 0; h < H_HEADS; ++h) {
    __syncthreads();
    for (int i = threadIdx.x; i < 1536; i += 256) {
      const int row = i >> 4, piece = (i & 15) * 8;
      *reinterpret_cast<float4*>(sWv + row * LS + piece) =
          *reinterpret_cast<const float4*>(Wv16 + (size_t)h * SLICE_H + row * 128 + piece);
    }
    __syncthreads();

    const float attn = att[(size_t)e * H_HEADS + h] /
                       (denom[(size_t)dstn * H_HEADS + h] + 1e-16f);
    v8f cv0 = {}, cv1 = {};
#pragma unroll
    for (int c = 0; c < 12; ++c) {
      const int tc = c >> 2, kc = (c & 3) * 32;
      if (__ballot(t == tc)) {
        v16h a_v = {};
        if (t == tc) a_v = av[c & 3];
        const int wb0 = (tc * 32 + col) * LS + kc + half * 8;
        const int wb1 = (tc * 32 + 16 + col) * LS + kc + half * 8;
        cv0 = __builtin_amdgcn_wmma_f32_16x16x32_f16(false, a_v, false, load_b(sWv + wb0),
                                                     (short)0, cv0, false, false);
        cv1 = __builtin_amdgcn_wmma_f32_16x16x32_f16(false, a_v, false, load_b(sWv + wb1),
                                                     (short)0, cv1, false, false);
      }
    }
#pragma unroll
    for (int r = 0; r < 8; ++r) {
      const int srow = r + hi8;
      const int srcl = (half << 4) + srow;
      const int   tr = __shfl(t, srcl, 32);
      const float ar = __shfl(attn, srcl, 32);
      const int   dn = __shfl(dstn, srcl, 32);
      const float v0 = (cv0[r] + bv[(h * 3 + tr) * 32 + col]) * ar;
      const float v1 = (cv1[r] + bv[(h * 3 + tr) * 32 + 16 + col]) * ar;
      atomicAdd(aggr + (size_t)dn * D_DIM + h * DKK + col, v0);
      atomicAdd(aggr + (size_t)dn * D_DIM + h * DKK + 16 + col, v1);
    }
  }
}

// ===================== K6: residual + exact GELU ===========================
__global__ void k_finish(const float* __restrict__ x, const float* __restrict__ aggr,
                         float* __restrict__ out) {
  const int i = blockIdx.x * blockDim.x + threadIdx.x;
  if (i >= N_NODES * D_DIM) return;
  const float a = aggr[i];
  out[i] = x[i] + 0.5f * a * (1.0f + erff(a * 0.7071067811865476f));
}

extern "C" void kernel_launch(void* const* d_in, const int* in_sizes, int n_in,
                              void* d_out, int out_size, void* d_ws, size_t ws_size,
                              hipStream_t stream) {
  const float* x     = (const float*)d_in[0];
  const int*   ei    = (const int*)d_in[1];
  const float* ev    = (const float*)d_in[2];
  // d_in[3] time_nodes: unused by reference
  const float* m1    = (const float*)d_in[5]; // diff_time_same_var -> t=1
  const float* m2    = (const float*)d_in[6]; // diff_time_diff_var -> t=2
  // d_in[7] n_layer: P==1 so always 0
  const float* gamma = (const float*)d_in[8];
  const float* beta  = (const float*)d_in[9];
  const float* Wq    = (const float*)d_in[10];
  const float* bq    = (const float*)d_in[11];
  const float* Wk    = (const float*)d_in[12];
  const float* bk    = (const float*)d_in[13];
  const float* Wv    = (const float*)d_in[14];
  const float* bv    = (const float*)d_in[15];
  float* out = (float*)d_out;

  char* w = (char*)d_ws;
  auto take = [&](size_t bytes) -> char* {
    char* p = w; w += (bytes + 255) & ~(size_t)255; return p;
  };
  _Float16* xn16 = (_Float16*)take((size_t)N_NODES * D_DIM * 2);
  _Float16* Wq16 = (_Float16*)take((size_t)W_HALVES * 2);
  _Float16* Wk16 = (_Float16*)take((size_t)W_HALVES * 2);
  _Float16* Wv16 = (_Float16*)take((size_t)W_HALVES * 2);
  float* att     = (float*)take((size_t)E_EDGES * H_HEADS * 4);
  float* denom   = (float*)take((size_t)N_NODES * H_HEADS * 4);
  unsigned* maxk = (unsigned*)take(256);
  float* aggr    = (float*)take((size_t)N_NODES * D_DIM * 4);

  hipMemsetAsync(denom, 0, (size_t)N_NODES * H_HEADS * 4, stream);
  hipMemsetAsync(maxk, 0, 4 * sizeof(unsigned), stream);
  hipMemsetAsync(aggr, 0, (size_t)N_NODES * D_DIM * 4, stream);

  k_layernorm<<<N_NODES / 8, 256, 0, stream>>>(x, gamma, beta, xn16);
  k_prep_w<<<(3 * W_HALVES + 255) / 256, 256, 0, stream>>>(Wq, Wk, Wv, Wq16, Wk16, Wv16);
  k_edge_qk<<<E_EDGES / (16 * 8), 256, 0, stream>>>(xn16, Wq16, Wk16, bq, bk, ei, ev,
                                                    m1, m2, att, maxk);
  k_softmax_denom<<<(E_EDGES * H_HEADS + 255) / 256, 256, 0, stream>>>(att, ei, maxk, denom);
  k_edge_v<<<E_EDGES / (16 * 8), 256, 0, stream>>>(xn16, Wv16, bv, ei, m1, m2, att, denom, aggr);
  k_finish<<<(N_NODES * D_DIM + 255) / 256, 256, 0, stream>>>(x, aggr, out);
}
